// TimeAwareFDR_75788992905501
// MI455X (gfx1250) — matrix operand order repaired and checked by reading, not compiled, not run on hardware
//
#include <hip/hip_runtime.h>
#include <hip/hip_bf16.h>

typedef __bf16 bf16_t;
typedef __attribute__((ext_vector_type(16))) __bf16 v16bf;
typedef __attribute__((ext_vector_type(8)))  float  v8f;
typedef __attribute__((ext_vector_type(4)))  int    v4i;

#define NN    1536
#define DD    512
#define HH    4
#define DHD   128
#define ASTR  40   // LDS row stride (elements) to avoid bank conflicts

#if defined(__AMDGCN__) && __has_builtin(__builtin_amdgcn_global_load_async_to_lds_b128) && __has_builtin(__builtin_amdgcn_s_wait_asynccnt)
#define USE_ASYNC_LDS 1
#else
#define USE_ASYNC_LDS 0
#endif

#define AS1 __attribute__((address_space(1)))
#define AS3 __attribute__((address_space(3)))

__device__ __forceinline__ bf16_t to_bf16(float f) { return (bf16_t)f; }

#if USE_ASYNC_LDS
__device__ __forceinline__ void async_copy16(const bf16_t* g, bf16_t* l)
{
    __builtin_amdgcn_global_load_async_to_lds_b128(
        (AS1 v4i*)(g), (AS3 v4i*)(l), 0, 0);
}
#endif

// ---------------------------------------------------------------------------
// Generic TN GEMM:  C[M,Nc] = alpha * (A[M,K] @ B[Nc,K]^T) + bias
// A,B bf16 row-major along K; C fp32 and/or bf16 outputs. Batched via z.
// 128x64 block tile, BK=32, 8 waves, each wave computes 32x32 via 4 WMMAs
// per K-step. Double-buffered LDS; async global->LDS staging when available.
// Requires M % 128 == 0 (true for all call sites: M = 1536).
// ---------------------------------------------------------------------------
__global__ __launch_bounds__(256)
void gemm_tn_bf16(const bf16_t* __restrict__ A, long aStride, int lda,
                  const bf16_t* __restrict__ B, long bStride, int ldb,
                  const float* __restrict__ bias,
                  float* __restrict__ C, bf16_t* __restrict__ Cb,
                  long cStride, int ldc,
                  int M, int Nc, int K, float alpha)
{
    __shared__ bf16_t As[2][128 * ASTR];
    __shared__ bf16_t Bs[2][64 * ASTR];

    const int bz = blockIdx.z;
    A += (long)bz * aStride;
    B += (long)bz * bStride;
    const long cOff = (long)bz * cStride;

    const int m0 = blockIdx.y * 128;
    const int n0 = blockIdx.x * 64;
    const int t    = threadIdx.x;
    const int wave = t >> 5;
    const int lane = t & 31;
    const int half = lane >> 4;
    const int l16  = lane & 15;
    const int rg   = wave & 3;   // row group: 32 rows each
    const int cg   = wave >> 2;  // col group: 32 cols each

    v8f acc00 = {}, acc01 = {}, acc10 = {}, acc11 = {};

    const int srow = t >> 2;         // 0..63
    const int scol = (t & 3) * 8;    // 0,8,16,24

    // stage one 128x32 A tile + 64x32 B tile into LDS buffer `buf`
    auto stage = [&](int buf, int k0) {
        int gr0 = m0 + srow;        if (gr0 >= M)  gr0 = M - 1;
        int gr1 = m0 + srow + 64;   if (gr1 >= M)  gr1 = M - 1;
        int gn  = n0 + srow;        if (gn  >= Nc) gn  = Nc - 1;
        const bf16_t* ga0 = A + (long)gr0 * lda + k0 + scol;
        const bf16_t* ga1 = A + (long)gr1 * lda + k0 + scol;
        const bf16_t* gb  = B + (long)gn  * ldb + k0 + scol;
        bf16_t* la0 = &As[buf][srow * ASTR + scol];
        bf16_t* la1 = &As[buf][(srow + 64) * ASTR + scol];
        bf16_t* lb  = &Bs[buf][srow * ASTR + scol];
#if USE_ASYNC_LDS
        async_copy16(ga0, la0);
        async_copy16(ga1, la1);
        async_copy16(gb,  lb);
#else
        *(uint4*)la0 = *(const uint4*)ga0;
        *(uint4*)la1 = *(const uint4*)ga1;
        *(uint4*)lb  = *(const uint4*)gb;
#endif
    };

    int buf = 0;
    stage(0, 0);

    for (int k0 = 0; k0 < K; k0 += 32) {
#if USE_ASYNC_LDS
        __builtin_amdgcn_s_wait_asynccnt(0);
#endif
        __syncthreads();

        if (k0 + 32 < K) stage(buf ^ 1, k0 + 32);
        if (k0 + 64 < K) {  // prefetch tile after next (global_prefetch_b8)
            __builtin_prefetch(A + (long)(m0 + srow) * lda + k0 + 64 + scol, 0, 1);
            __builtin_prefetch(B + (long)(n0 + (srow & 63)) * ldb + k0 + 64 + scol, 0, 1);
        }

        // ---- fragments (ISA 16-bit A 16x32 / B 32x16 layouts) ----
        union Frag { v16bf v; uint4 q[2]; } a0, a1, b0, b1;
        const bf16_t* Ab = As[buf];
        const bf16_t* Bb = Bs[buf];

        const bf16_t* ap0 = &Ab[(rg * 32 + l16) * ASTR];
        a0.q[0] = *(const uint4*)(ap0 + half * 8);
        a0.q[1] = *(const uint4*)(ap0 + 16 + half * 8);
        const bf16_t* ap1 = &Ab[(rg * 32 + 16 + l16) * ASTR];
        a1.q[0] = *(const uint4*)(ap1 + half * 8);
        a1.q[1] = *(const uint4*)(ap1 + 16 + half * 8);

        const bf16_t* bp0 = &Bb[(cg * 32 + l16) * ASTR + half * 8];
        b0.q[0] = *(const uint4*)(bp0);
        b0.q[1] = *(const uint4*)(bp0 + 16);
        const bf16_t* bp1 = &Bb[(cg * 32 + 16 + l16) * ASTR + half * 8];
        b1.q[0] = *(const uint4*)(bp1);
        b1.q[1] = *(const uint4*)(bp1 + 16);

        acc00 = __builtin_amdgcn_wmma_f32_16x16x32_bf16(false, a0.v, false, b0.v, (short)0, acc00, false, false);
        acc01 = __builtin_amdgcn_wmma_f32_16x16x32_bf16(false, a0.v, false, b1.v, (short)0, acc01, false, false);
        acc10 = __builtin_amdgcn_wmma_f32_16x16x32_bf16(false, a1.v, false, b0.v, (short)0, acc10, false, false);
        acc11 = __builtin_amdgcn_wmma_f32_16x16x32_bf16(false, a1.v, false, b1.v, (short)0, acc11, false, false);

        buf ^= 1;
    }

    // ---- epilogue (C layout: VGPR r -> row base + half*8 + r, col = l16) ----
    const int cm0 = m0 + rg * 32 + half * 8;
    const int cm1 = cm0 + 16;
    const int cn0 = n0 + cg * 32 + l16;
    const int cn1 = cn0 + 16;
    const float bv0 = (bias && cn0 < Nc) ? bias[cn0] : 0.0f;
    const float bv1 = (bias && cn1 < Nc) ? bias[cn1] : 0.0f;

    #pragma unroll
    for (int r = 0; r < 8; ++r) {
        const int r0 = cm0 + r, r1 = cm1 + r;
        const float v00 = acc00[r] * alpha + bv0;
        const float v01 = acc01[r] * alpha + bv1;
        const float v10 = acc10[r] * alpha + bv0;
        const float v11 = acc11[r] * alpha + bv1;
        if (r0 < M) {
            const long base = cOff + (long)r0 * ldc;
            if (cn0 < Nc) { if (C) C[base + cn0] = v00; if (Cb) Cb[base + cn0] = to_bf16(v00); }
            if (cn1 < Nc) { if (C) C[base + cn1] = v01; if (Cb) Cb[base + cn1] = to_bf16(v01); }
        }
        if (r1 < M) {
            const long base = cOff + (long)r1 * ldc;
            if (cn0 < Nc) { if (C) C[base + cn0] = v10; if (Cb) Cb[base + cn0] = to_bf16(v10); }
            if (cn1 < Nc) { if (C) C[base + cn1] = v11; if (Cb) Cb[base + cn1] = to_bf16(v11); }
        }
    }
}

// ---------------------------------------------------------------------------
// Elementwise / reduction helper kernels
// ---------------------------------------------------------------------------
__global__ void k_build_x(const float* __restrict__ emb, const float* __restrict__ tp,
                          const float* __restrict__ wt, const float* __restrict__ bt,
                          bf16_t* __restrict__ xb)
{
    long i = (long)blockIdx.x * 256 + threadIdx.x;
    if (i >= (long)NN * DD) return;
    int col = (int)(i & (DD - 1));
    long row = i >> 9;
    xb[i] = to_bf16(emb[i] + tp[row] * wt[col] + bt[col]);
}

__global__ void k_convert_crop(const float* __restrict__ src, bf16_t* __restrict__ dst,
                               int rows, int srcld, int cols)
{
    long i = (long)blockIdx.x * 256 + threadIdx.x;
    if (i >= (long)rows * cols) return;
    long r = i / cols;
    int  c = (int)(i - r * cols);
    dst[i] = to_bf16(src[r * srcld + c]);
}

__global__ void k_transpose_v(const bf16_t* __restrict__ qkv, bf16_t* __restrict__ vT)
{
    // vT[h*128+d][q] = qkv[q][1024 + h*128 + d]
    long i = (long)blockIdx.x * 256 + threadIdx.x;
    if (i >= (long)DD * NN) return;
    int  q  = (int)(i % NN);
    long hd = i / NN;
    vT[i] = qkv[(long)q * (3 * DD) + 2 * DD + hd];
}

__global__ __launch_bounds__(256)
void k_softmax_rows(float* __restrict__ S, bf16_t* __restrict__ Sb,
                    long batchStride, int len)
{
    const int row = blockIdx.x, b = blockIdx.y, t = threadIdx.x;
    float* p = S + (long)b * batchStride + (long)row * len;
    __shared__ float red[256];

    float mx = -3.4e38f;
    for (int i = t; i < len; i += 256) mx = fmaxf(mx, p[i]);
    red[t] = mx; __syncthreads();
    for (int s = 128; s > 0; s >>= 1) { if (t < s) red[t] = fmaxf(red[t], red[t + s]); __syncthreads(); }
    mx = red[0]; __syncthreads();

    float sum = 0.0f;
    for (int i = t; i < len; i += 256) { float e = __expf(p[i] - mx); p[i] = e; sum += e; }
    red[t] = sum; __syncthreads();
    for (int s = 128; s > 0; s >>= 1) { if (t < s) red[t] += red[t + s]; __syncthreads(); }
    const float inv = 1.0f / red[0];

    for (int i = t; i < len; i += 256) {
        float v = p[i] * inv;
        p[i] = v;
        if (Sb) Sb[(long)b * batchStride + (long)row * len + i] = to_bf16(v);
    }
}

__global__ void k_colsum(const float* __restrict__ P, float* __restrict__ out, int n, int batch)
{
    int k = blockIdx.x * 256 + threadIdx.x;
    if (k >= n) return;
    float s = 0.0f;
    for (int b = 0; b < batch; ++b) {
        const float* p = P + (long)b * n * n + k;
        for (int q = 0; q < n; ++q) s += p[(long)q * n];
    }
    out[k] = s;
}

__global__ void k_time_grid(const float* __restrict__ tp, int n,
                            float* __restrict__ startv, float* __restrict__ endv)
{
    __shared__ float smn[256], smx[256];
    int t = threadIdx.x;
    float mn = 3.4e38f, mx = -3.4e38f;
    for (int i = t; i < n; i += 256) { float v = tp[i]; mn = fminf(mn, v); mx = fmaxf(mx, v); }
    smn[t] = mn; smx[t] = mx; __syncthreads();
    for (int s = 128; s > 0; s >>= 1) {
        if (t < s) { smn[t] = fminf(smn[t], smn[t + s]); smx[t] = fmaxf(smx[t], smx[t + s]); }
        __syncthreads();
    }
    if (t < 5) {
        float a = smn[0], step = (smx[0] - smn[0]) / 5.0f;
        startv[t] = a + step * t;
        endv[t]   = a + step * (t + 1);
    }
}

__global__ void k_precompute_qvec(const float* __restrict__ in_w, const float* __restrict__ in_b,
                                  const float* __restrict__ qp_w, const float* __restrict__ qp_b,
                                  float* __restrict__ u_s, float* __restrict__ u_e,
                                  float* __restrict__ c0)
{
    int d = blockIdx.x * 256 + threadIdx.x;
    if (d >= DD) return;
    const float* wr = in_w + (long)d * DD;   // wq row d
    float s = 0.f, e = 0.f, c = 0.f;
    for (int j = 0; j < DD; ++j) {
        float w = wr[j];
        s += w * qp_w[(long)j * (DD + 2) + DD];
        e += w * qp_w[(long)j * (DD + 2) + DD + 1];
        c += w * qp_b[j];
    }
    u_s[d] = s; u_e[d] = e; c0[d] = c + in_b[d];
}

__global__ void k_pooled_mean(const float* __restrict__ attn, float* __restrict__ pooled, int n)
{
    int d = blockIdx.x * 256 + threadIdx.x;
    if (d >= DD) return;
    float s = 0.f;
    for (int q = 0; q < n; ++q) s += attn[(long)q * DD + d];
    pooled[d] = s / (float)n;
}

__global__ __launch_bounds__(128)
void k_ic_head(const float* __restrict__ pooled,
               const float* __restrict__ w1, const float* __restrict__ b1,
               const float* __restrict__ w2, const float* __restrict__ b2,
               float* __restrict__ outMask)
{
    __shared__ float h[128];
    __shared__ float lg[5];
    int t = threadIdx.x;
    if (t < 128) {
        float s = b1[t];
        const float* wr = w1 + (long)t * DD;
        for (int d = 0; d < DD; ++d) s += wr[d] * pooled[d];
        h[t] = fmaxf(s, 0.0f);
    }
    __syncthreads();
    if (t < 5) {
        float s = b2[t];
        const float* wr = w2 + (long)t * 128;
        for (int r = 0; r < 128; ++r) s += wr[r] * h[r];
        lg[t] = s;
    }
    __syncthreads();
    if (t == 0) {
        int best = 0;
        for (int i = 1; i < 5; ++i) if (lg[i] > lg[best]) best = i;
        int num = best + 1;
        for (int i = 0; i < 5; ++i) outMask[i] = (i < num) ? 1.0f : 0.0f;
    }
}

__global__ void k_build_q2(const float* __restrict__ q2base,
                           const float* __restrict__ us, const float* __restrict__ ue,
                           const float* __restrict__ c0,
                           const float* __restrict__ startv, const float* __restrict__ endv,
                           int ii, bf16_t* __restrict__ out)
{
    long i = (long)blockIdx.x * 256 + threadIdx.x;
    if (i >= (long)NN * DD) return;
    int col = (int)(i & (DD - 1));
    float s = startv[ii], e = endv[ii];
    out[i] = to_bf16(q2base[i] + s * us[col] + e * ue[col] + c0[col]);
}

__global__ void k_build_h(const float* __restrict__ h1base,
                          const float* __restrict__ ref_w1,  // pre-offset layer, [128,514] f32
                          const float* __restrict__ ref_b1,  // pre-offset layer
                          const float* __restrict__ startv, const float* __restrict__ endv,
                          int ii, bf16_t* __restrict__ out)
{
    long i = (long)blockIdx.x * 256 + threadIdx.x;
    if (i >= (long)NN * 128) return;
    int col = (int)(i & 127);
    float s = startv[ii], e = endv[ii];
    float w0 = ref_w1[(long)col * (DD + 2) + DD];
    float w1 = ref_w1[(long)col * (DD + 2) + DD + 1];
    float v = h1base[i] + s * w0 + e * w1 + ref_b1[col];
    out[i] = to_bf16(fmaxf(v, 0.0f));
}

__global__ __launch_bounds__(128)
void k_soft_off(const float* __restrict__ logits, const float* __restrict__ wparams,
                float* __restrict__ s_off, float* __restrict__ e_off)
{
    const int row = blockIdx.x, t = threadIdx.x;
    const float* p = logits + (long)row * 200;
    __shared__ float red[128];
    for (int hsel = 0; hsel < 2; ++hsel) {
        const float* q = p + hsel * 100;
        float mx = (t < 100) ? q[t] : -3.4e38f;
        red[t] = mx; __syncthreads();
        for (int s = 64; s > 0; s >>= 1) { if (t < s) red[t] = fmaxf(red[t], red[t + s]); __syncthreads(); }
        mx = red[0]; __syncthreads();
        float e = (t < 100) ? __expf(q[t] - mx) : 0.0f;
        red[t] = e; __syncthreads();
        for (int s = 64; s > 0; s >>= 1) { if (t < s) red[t] += red[t + s]; __syncthreads(); }
        float denom = red[0]; __syncthreads();
        red[t] = (t < 100) ? e * wparams[t] : 0.0f; __syncthreads();
        for (int s = 64; s > 0; s >>= 1) { if (t < s) red[t] += red[t + s]; __syncthreads(); }
        float val = red[0] / denom;
        if (t == 0) { if (hsel == 0) s_off[row] = val; else e_off[row] = val; }
        __syncthreads();
    }
}

__global__ __launch_bounds__(256)
void k_delta_reduce(const float* __restrict__ colsumH,
                    const float* __restrict__ s_off, const float* __restrict__ e_off,
                    float* __restrict__ dds, float* __restrict__ dde, int ii, int n)
{
    __shared__ float r1[256], r2[256];
    int t = threadIdx.x;
    float a = 0.f, b = 0.f;
    for (int k = t; k < n; k += 256) { float c = colsumH[k]; a += c * s_off[k]; b += c * e_off[k]; }
    r1[t] = a; r2[t] = b; __syncthreads();
    for (int s = 128; s > 0; s >>= 1) { if (t < s) { r1[t] += r1[t + s]; r2[t] += r2[t + s]; } __syncthreads(); }
    if (t == 0) { dds[ii] = r1[0] * 0.25f; dde[ii] = r2[0] * 0.25f; }  // mean over H=4 heads
}

__global__ void k_apply_delta(float* __restrict__ startv, float* __restrict__ endv,
                              const float* __restrict__ dds, const float* __restrict__ dde)
{
    int i = threadIdx.x;
    if (i < 5) { startv[i] += dds[i]; endv[i] += dde[i]; }
}

__global__ void k_write_bounds(const float* __restrict__ startv, const float* __restrict__ endv,
                               float* __restrict__ out)
{
    int i = threadIdx.x;
    if (i < 5) { out[2 * i] = startv[i]; out[2 * i + 1] = endv[i]; }
}

// ---------------------------------------------------------------------------
extern "C" void kernel_launch(void* const* d_in, const int* in_sizes, int n_in,
                              void* d_out, int out_size, void* d_ws, size_t ws_size,
                              hipStream_t stream)
{
    const float* emb    = (const float*)d_in[0];
    const float* tp     = (const float*)d_in[1];
    const float* W_time = (const float*)d_in[3];
    const float* b_time = (const float*)d_in[4];
    const float* in_w   = (const float*)d_in[5];
    const float* in_b   = (const float*)d_in[6];
    const float* out_w  = (const float*)d_in[7];
    const float* out_b  = (const float*)d_in[8];
    const float* ic_w1  = (const float*)d_in[9];
    const float* ic_b1  = (const float*)d_in[10];
    const float* ic_w2  = (const float*)d_in[11];
    const float* ic_b2  = (const float*)d_in[12];
    const float* ref_w1 = (const float*)d_in[13];
    const float* ref_b1 = (const float*)d_in[14];
    const float* ref_w2 = (const float*)d_in[15];
    const float* ref_b2 = (const float*)d_in[16];
    const float* wparams= (const float*)d_in[17];
    const float* qp_w   = (const float*)d_in[18];
    const float* qp_b   = (const float*)d_in[19];
    float* outF = (float*)d_out;

    // ---- workspace arena ----
    char* ws = (char*)d_ws;
    size_t off = 0;
    auto alloc = [&](size_t bytes) -> char* {
        char* p = ws + off;
        off += (bytes + 255) & ~(size_t)255;
        return p;
    };
    bf16_t* x_bf       = (bf16_t*)alloc((size_t)NN * DD * 2);
    bf16_t* w_in_bf    = (bf16_t*)alloc((size_t)3 * DD * DD * 2);
    bf16_t* w_out_bf   = (bf16_t*)alloc((size_t)DD * DD * 2);
    bf16_t* qp_w_bf    = (bf16_t*)alloc((size_t)DD * DD * 2);
    bf16_t* ref_w1_bf  = (bf16_t*)alloc((size_t)3 * 128 * DD * 2);
    bf16_t* ref_w2_bf  = (bf16_t*)alloc((size_t)3 * 200 * 128 * 2);
    bf16_t* qkv_bf     = (bf16_t*)alloc((size_t)NN * 3 * DD * 2);
    bf16_t* vT_bf      = (bf16_t*)alloc((size_t)DD * NN * 2);
    float*  probs      = (float*) alloc((size_t)HH * NN * NN * 4);
    bf16_t* probs_bf   = (bf16_t*)alloc((size_t)HH * NN * NN * 2);
    float*  attn_h     = (float*) alloc((size_t)NN * DD * 4);
    bf16_t* attn_h_bf  = (bf16_t*)alloc((size_t)NN * DD * 2);
    float*  attn_out   = (float*) alloc((size_t)NN * DD * 4);
    bf16_t* attn_o_bf  = (bf16_t*)alloc((size_t)NN * DD * 2);
    bf16_t* k2_bf      = (bf16_t*)alloc((size_t)NN * DD * 2);
    bf16_t* qp_base_bf = (bf16_t*)alloc((size_t)NN * DD * 2);
    float*  q2_base    = (float*) alloc((size_t)NN * DD * 4);
    bf16_t* q2_bf      = (bf16_t*)alloc((size_t)NN * DD * 2);
    float*  h1_base    = (float*) alloc((size_t)NN * 128 * 4);
    bf16_t* h_bf       = (bf16_t*)alloc((size_t)NN * 128 * 2);
    float*  logitsB    = (float*) alloc((size_t)NN * 200 * 4);
    float*  colsumH    = (float*) alloc((size_t)NN * 4);
    float*  s_off      = (float*) alloc((size_t)NN * 4);
    float*  e_off      = (float*) alloc((size_t)NN * 4);
    float*  u_s        = (float*) alloc(DD * 4);
    float*  u_e        = (float*) alloc(DD * 4);
    float*  c0         = (float*) alloc(DD * 4);
    float*  pooled     = (float*) alloc(DD * 4);
    float*  startv     = (float*) alloc(5 * 4);
    float*  endv       = (float*) alloc(5 * 4);
    float*  dds        = (float*) alloc(5 * 4);
    float*  dde        = (float*) alloc(5 * 4);
    (void)ws_size; (void)n_in; (void)in_sizes; (void)out_size;

    const float scale = 0.088388347648318447f;  // 1/sqrt(128)
    const dim3 B256(256), B128(128);

    // 1) x = emb + t * W_time^T + b  (bf16)
    k_build_x<<<(NN * DD + 255) / 256, B256, 0, stream>>>(emb, tp, W_time, b_time, x_bf);

    // 2) weight conversions (crop D+2 -> D where applicable)
    k_convert_crop<<<(3 * DD * DD + 255) / 256, B256, 0, stream>>>(in_w,  w_in_bf,  3 * DD, DD, DD);
    k_convert_crop<<<(DD * DD + 255) / 256,     B256, 0, stream>>>(out_w, w_out_bf, DD, DD, DD);
    k_convert_crop<<<(DD * DD + 255) / 256,     B256, 0, stream>>>(qp_w,  qp_w_bf,  DD, DD + 2, DD);
    k_convert_crop<<<(3 * 128 * DD + 255) / 256,B256, 0, stream>>>(ref_w1, ref_w1_bf, 3 * 128, DD + 2, DD);
    k_convert_crop<<<(3 * 200 * 128 + 255)/256, B256, 0, stream>>>(ref_w2, ref_w2_bf, 3 * 200, 128, 128);

    // 3) qkv = x @ in_w^T + in_b   [N,1536] bf16
    gemm_tn_bf16<<<dim3(24, 12, 1), B256, 0, stream>>>(
        x_bf, 0, DD, w_in_bf, 0, DD, in_b, nullptr, qkv_bf, 0, 3 * DD,
        NN, 3 * DD, DD, 1.0f);

    // 4) vT[h][d][q]
    k_transpose_v<<<(DD * NN + 255) / 256, B256, 0, stream>>>(qkv_bf, vT_bf);

    // 5) main scores: S_h = q_h @ k_h^T * scale   (batched over 4 heads)
    gemm_tn_bf16<<<dim3(24, 12, HH), B256, 0, stream>>>(
        qkv_bf, DHD, 3 * DD, qkv_bf + DD, DHD, 3 * DD, nullptr,
        probs, nullptr, (long)NN * NN, NN, NN, NN, DHD, scale);

    // 6) row softmax (also bf16 for PV)
    k_softmax_rows<<<dim3(NN, HH), B256, 0, stream>>>(probs, probs_bf, (long)NN * NN, NN);

    // 7) attn_heads[:, h*128:] = P_h @ V_h
    gemm_tn_bf16<<<dim3(2, 12, HH), B256, 0, stream>>>(
        probs_bf, (long)NN * NN, NN, vT_bf, (long)DHD * NN, NN, nullptr,
        attn_h, attn_h_bf, DHD, DD, NN, DHD, NN, 1.0f);

    // 8) attn_out = attn_heads @ out_w^T + out_b
    gemm_tn_bf16<<<dim3(8, 12, 1), B256, 0, stream>>>(
        attn_h_bf, 0, DD, w_out_bf, 0, DD, out_b,
        attn_out, attn_o_bf, 0, DD, NN, DD, DD, 1.0f);

    // 9) pooled / interval-count head -> mask in d_out[10..14]
    k_pooled_mean<<<2, B256, 0, stream>>>(attn_out, pooled, NN);
    k_ic_head<<<1, B128, 0, stream>>>(pooled, ic_w1, ic_b1, ic_w2, ic_b2, outF + 10);

    // 10) time grid -> start/end
    k_time_grid<<<1, B256, 0, stream>>>(tp, NN, startv, endv);

    // 11) rank-2 decomposition vectors: u_s = wq@qp_w[:,512], u_e, c0 = wq@qp_b + bq
    k_precompute_qvec<<<2, B256, 0, stream>>>(in_w, in_b, qp_w, qp_b, u_s, u_e, c0);

    // 12) k2 = attn_out @ wk^T + bk  (interval/layer invariant)
    gemm_tn_bf16<<<dim3(8, 12, 1), B256, 0, stream>>>(
        attn_o_bf, 0, DD, w_in_bf + (size_t)DD * DD, 0, DD, in_b + DD,
        nullptr, k2_bf, 0, DD, NN, DD, DD, 1.0f);

    // 13) qp_base = attn_out @ qp_w[:, :512]^T
    gemm_tn_bf16<<<dim3(8, 12, 1), B256, 0, stream>>>(
        attn_o_bf, 0, DD, qp_w_bf, 0, DD, nullptr,
        nullptr, qp_base_bf, 0, DD, NN, DD, DD, 1.0f);

    // 14) q2_base = qp_base @ wq^T
    gemm_tn_bf16<<<dim3(8, 12, 1), B256, 0, stream>>>(
        qp_base_bf, 0, DD, w_in_bf, 0, DD, nullptr,
        q2_base, nullptr, 0, DD, NN, DD, DD, 1.0f);

    // ---- 3 refinement layers ----
    for (int l = 0; l < 3; ++l) {
        // h1_base = attn_out @ ref_w1[l][:, :512]^T
        gemm_tn_bf16<<<dim3(2, 12, 1), B256, 0, stream>>>(
            attn_o_bf, 0, DD, ref_w1_bf + (size_t)l * 128 * DD, 0, DD, nullptr,
            h1_base, nullptr, 0, 128, NN, 128, DD, 1.0f);

        for (int ii = 0; ii < 5; ++ii) {
            // q2 = q2_base + s*u_s + e*u_e + c0  (bf16)
            k_build_q2<<<(NN * DD + 255) / 256, B256, 0, stream>>>(
                q2_base, u_s, u_e, c0, startv, endv, ii, q2_bf);

            // inner scores: q2_h @ k2_h^T * scale, batched heads
            gemm_tn_bf16<<<dim3(24, 12, HH), B256, 0, stream>>>(
                q2_bf, DHD, DD, k2_bf, DHD, DD, nullptr,
                probs, nullptr, (long)NN * NN, NN, NN, NN, DHD, scale);

            k_softmax_rows<<<dim3(NN, HH), B256, 0, stream>>>(probs, nullptr, (long)NN * NN, NN);
            k_colsum<<<(NN + 255) / 256, B256, 0, stream>>>(probs, colsumH, NN, HH);

            // h = relu(h1_base + s*w1[:,512] + e*w1[:,513] + b1)
            k_build_h<<<(NN * 128 + 255) / 256, B256, 0, stream>>>(
                h1_base, ref_w1 + (size_t)l * 128 * (DD + 2), ref_b1 + (size_t)l * 128,
                startv, endv, ii, h_bf);

            // logits = h @ ref_w2[l]^T + ref_b2[l]   [N,200]
            gemm_tn_bf16<<<dim3(4, 12, 1), B256, 0, stream>>>(
                h_bf, 0, 128, ref_w2_bf + (size_t)l * 200 * 128, 0, 128,
                ref_b2 + (size_t)l * 200, logitsB, nullptr, 0, 200, NN, 200, 128, 1.0f);

            k_soft_off<<<NN, B128, 0, stream>>>(logitsB, wparams, s_off, e_off);
            k_delta_reduce<<<1, B256, 0, stream>>>(colsumH, s_off, e_off, dds, dde, ii, NN);
        }
        k_apply_delta<<<1, 32, 0, stream>>>(startv, endv, dds, dde);
    }

    // final bounds -> d_out[0..9]
    k_write_bounds<<<1, 32, 0, stream>>>(startv, endv, outF);
}
